// GLIFR_36764920053992
// MI455X (gfx1250) — matrix operand order repaired and checked by reading, not compiled, not run on hardware
//
#include <hip/hip_runtime.h>

// ---------------------------------------------------------------------------
// GLIFR step for MI455X (gfx1250, wave32, WMMA).
//   syncurrent = x @ W_iv + firing_delayed @ W_lat
//   bf16 hi/lo split (3-term) WMMA with fp32 accumulate (~fp32 accuracy),
//   weights/activations pre-packed into WMMA fragment layout, async-LDS
//   double-buffered pipeline. Elementwise GLIF update fused in 2nd kernel.
// ---------------------------------------------------------------------------

typedef __bf16 bf16;
typedef __attribute__((ext_vector_type(16))) __bf16 v16bf;
typedef __attribute__((ext_vector_type(8)))  __bf16 bf16x8;
typedef __attribute__((ext_vector_type(4)))  __bf16 bf16x4;
typedef __attribute__((ext_vector_type(8)))  float   v8f;
typedef __attribute__((ext_vector_type(4)))  float   f32x4;

#define BDIM   2048
#define INDIM  2048
#define HDIM   2048
#define BH     (BDIM * HDIM)

#define DT_C   0.05f
#define TAU_C  0.05f
#define R_C    0.1f
#define I0_C   0.7f

#define BM 128
#define BN 128
#define BK 32

// packed matrix: [kt(64)][tile(128)][chunk(2)][lane(32)][8 bf16]; 512 bf16/tile
#define PKMAT  ((size_t)64 * 128 * 512)   // bf16 elements per packed matrix half

union Frag { v16bf v; bf16x8 h[2]; };

__device__ __forceinline__ float sigf(float x) {
  return 1.0f / (1.0f + __expf(-x));
}

// ---------------------------------------------------------------------------
// async global -> LDS copy of 32 bytes per thread (2 x b128)
// builtin signature (from diagnostics): (v4i AS1*, v4i AS3*, imm, imm)
// ---------------------------------------------------------------------------
#if __has_builtin(__builtin_amdgcn_global_load_async_to_lds_b128)
#define USE_ASYNC 1
typedef int v4i __attribute__((__vector_size__(16)));
typedef __attribute__((address_space(1))) v4i g_v4i;
typedef __attribute__((address_space(3))) v4i l_v4i;
#else
#define USE_ASYNC 0
#endif

__device__ __forceinline__ void cp32B(const bf16* g, bf16* l) {
#if USE_ASYNC
  __builtin_amdgcn_global_load_async_to_lds_b128((g_v4i*)g,       (l_v4i*)l,       0, 0);
  __builtin_amdgcn_global_load_async_to_lds_b128((g_v4i*)(g + 8), (l_v4i*)(l + 8), 0, 0);
#else
  *(uint4*)l       = *(const uint4*)g;
  *(uint4*)(l + 8) = *(const uint4*)(g + 8);
#endif
}

__device__ __forceinline__ void wait_async_prev() {   // older batch (8) done
#if USE_ASYNC
#if __has_builtin(__builtin_amdgcn_s_wait_asynccnt)
  __builtin_amdgcn_s_wait_asynccnt(8);
#else
  asm volatile("s_wait_asynccnt 0x8" ::: "memory");
#endif
#endif
}

__device__ __forceinline__ void wait_async_all() {
#if USE_ASYNC
#if __has_builtin(__builtin_amdgcn_s_wait_asynccnt)
  __builtin_amdgcn_s_wait_asynccnt(0);
#else
  asm volatile("s_wait_asynccnt 0x0" ::: "memory");
#endif
#endif
}

// ---------------------------------------------------------------------------
// Pack kernels: fp32 matrix -> bf16 hi/lo pair in WMMA fragment order.
// One wave packs one 16x32 (A) or 32x16 (B) tile.
// A-fragment layout: lane l=(half,r): elem e -> K=half*16+e
// B-fragment layout: lane l=(khalf,n): elem e -> K=khalf*16+e
// ---------------------------------------------------------------------------
__global__ __launch_bounds__(256)
void pack_a(const float* __restrict__ src, bf16* __restrict__ dhi,
            bf16* __restrict__ dlo) {
  const int wid  = blockIdx.x * 8 + (threadIdx.x >> 5);  // tile id: kt*128+mt
  const int l    = threadIdx.x & 31;
  const int kt   = wid >> 7;
  const int mt   = wid & 127;
  const int half = l >> 4, r = l & 15;
  const float* row = src + (size_t)(mt * 16 + r) * INDIM + kt * 32;
  const size_t tb = (size_t)wid * 512;
#pragma unroll
  for (int c = 0; c < 2; ++c) {
    const int kb = half * 16 + c * 8;
    f32x4 a = *(const f32x4*)(row + kb);
    f32x4 b = *(const f32x4*)(row + kb + 4);
    bf16x8 h, lo;
#pragma unroll
    for (int e = 0; e < 4; ++e) {
      bf16 hh = (bf16)a[e];
      h[e] = hh;  lo[e] = (bf16)(a[e] - (float)hh);
      bf16 hb = (bf16)b[e];
      h[4 + e] = hb;  lo[4 + e] = (bf16)(b[e] - (float)hb);
    }
    *(bf16x8*)(dhi + tb + c * 256 + l * 8) = h;
    *(bf16x8*)(dlo + tb + c * 256 + l * 8) = lo;
  }
}

__global__ __launch_bounds__(256)
void pack_b(const float* __restrict__ src, bf16* __restrict__ dhi,
            bf16* __restrict__ dlo) {
  const int wid = blockIdx.x * 8 + (threadIdx.x >> 5);   // tile id: kt*128+nt
  const int l   = threadIdx.x & 31;
  const int kt  = wid >> 7;
  const int nt  = wid & 127;
  const int n   = l & 15, kh = l >> 4;
  const float* col = src + (size_t)(kt * 32 + kh * 16) * HDIM + nt * 16 + n;
  const size_t tb = (size_t)wid * 512;
#pragma unroll
  for (int c = 0; c < 2; ++c) {
    bf16x8 h, lo;
#pragma unroll
    for (int e = 0; e < 8; ++e) {
      float v = col[(size_t)(c * 8 + e) * HDIM];
      bf16 hh = (bf16)v;
      h[e] = hh;  lo[e] = (bf16)(v - (float)hh);
    }
    *(bf16x8*)(dhi + tb + c * 256 + l * 8) = h;
    *(bf16x8*)(dlo + tb + c * 256 + l * 8) = lo;
  }
}

// ---------------------------------------------------------------------------
// Packed-input GEMM: async-LDS double-buffered, 24 WMMA per K-step.
// ---------------------------------------------------------------------------
__global__ __launch_bounds__(256)
void glifr_gemm_pk(const bf16* __restrict__ xhi,   const bf16* __restrict__ xlo,
                   const bf16* __restrict__ fdhi,  const bf16* __restrict__ fdlo,
                   const bf16* __restrict__ wivhi, const bf16* __restrict__ wivlo,
                   const bf16* __restrict__ wlathi,const bf16* __restrict__ wlatlo,
                   float* __restrict__ syn) {
  __shared__ bf16 L[2][4][4096];   // 64 KB: [stage][Ahi,Alo,Bhi,Blo][8 tiles*512]

  const int t    = threadIdx.x;
  const int lane = t & 31;
  const int half = lane >> 4;
  const int r    = lane & 15;
  const int w    = t >> 5, wm = w >> 2, wn = w & 3;
  const int mt0  = blockIdx.y * 8;
  const int nt0  = blockIdx.x * 8;

  v8f acc[4][2] = {};

  // prologue: stage 0 <- kt 0 (stream 0: x / W_iv)
  {
    const size_t ao = ((size_t)mt0) * 512 + t * 16;
    const size_t bo = ((size_t)nt0) * 512 + t * 16;
    cp32B(xhi   + ao, &L[0][0][t * 16]);
    cp32B(xlo   + ao, &L[0][1][t * 16]);
    cp32B(wivhi + bo, &L[0][2][t * 16]);
    cp32B(wivlo + bo, &L[0][3][t * 16]);
  }

  for (int kt = 0; kt < 128; ++kt) {
    const int st = kt & 1;
    if (kt + 1 < 128) {
      const int k2 = kt + 1;
      const int s  = k2 >> 6, kk = k2 & 63, ns = st ^ 1;
      const bf16* Ah = s ? fdhi   : xhi;
      const bf16* Al = s ? fdlo   : xlo;
      const bf16* Bh = s ? wlathi : wivhi;
      const bf16* Bl = s ? wlatlo : wivlo;
      const size_t ao = ((size_t)kk * 128 + mt0) * 512 + t * 16;
      const size_t bo = ((size_t)kk * 128 + nt0) * 512 + t * 16;
      cp32B(Ah + ao, &L[ns][0][t * 16]);
      cp32B(Al + ao, &L[ns][1][t * 16]);
      cp32B(Bh + bo, &L[ns][2][t * 16]);
      cp32B(Bl + bo, &L[ns][3][t * 16]);
      wait_async_prev();
    } else {
      wait_async_all();
    }
    __syncthreads();   // current stage resident for all waves

    Frag fah[4], fal[4], fbh[2], fbl[2];
#pragma unroll
    for (int mi = 0; mi < 4; ++mi) {
      const bf16* p = &L[st][0][(wm * 4 + mi) * 512 + lane * 8];
      fah[mi].h[0] = *(const bf16x8*)p;
      fah[mi].h[1] = *(const bf16x8*)(p + 256);
      const bf16* q = &L[st][1][(wm * 4 + mi) * 512 + lane * 8];
      fal[mi].h[0] = *(const bf16x8*)q;
      fal[mi].h[1] = *(const bf16x8*)(q + 256);
    }
#pragma unroll
    for (int ni = 0; ni < 2; ++ni) {
      const bf16* p = &L[st][2][(wn * 2 + ni) * 512 + lane * 8];
      fbh[ni].h[0] = *(const bf16x8*)p;
      fbh[ni].h[1] = *(const bf16x8*)(p + 256);
      const bf16* q = &L[st][3][(wn * 2 + ni) * 512 + lane * 8];
      fbl[ni].h[0] = *(const bf16x8*)q;
      fbl[ni].h[1] = *(const bf16x8*)(q + 256);
    }

#pragma unroll
    for (int mi = 0; mi < 4; ++mi)
#pragma unroll
      for (int ni = 0; ni < 2; ++ni) {
        acc[mi][ni] = __builtin_amdgcn_wmma_f32_16x16x32_bf16(
            false, fah[mi].v, false, fbh[ni].v, (short)0, acc[mi][ni], false, false);
        acc[mi][ni] = __builtin_amdgcn_wmma_f32_16x16x32_bf16(
            false, fal[mi].v, false, fbh[ni].v, (short)0, acc[mi][ni], false, false);
        acc[mi][ni] = __builtin_amdgcn_wmma_f32_16x16x32_bf16(
            false, fah[mi].v, false, fbl[ni].v, (short)0, acc[mi][ni], false, false);
      }
    __syncthreads();   // all waves done before this stage is overwritten
  }

  // C layout: VGPR v, lane l -> row = half*8+v, col = r
#pragma unroll
  for (int mi = 0; mi < 4; ++mi)
#pragma unroll
    for (int ni = 0; ni < 2; ++ni) {
      const int row = blockIdx.y * BM + wm * 64 + mi * 16 + half * 8;
      const int col = blockIdx.x * BN + wn * 32 + ni * 16 + r;
#pragma unroll
      for (int v = 0; v < 8; ++v)
        syn[(size_t)(row + v) * HDIM + col] = acc[mi][ni][v];
    }
}

// ---------------------------------------------------------------------------
// Fallback fused GEMM (used only if d_ws is too small for packing).
// ---------------------------------------------------------------------------
#define ASTR 40
#define BSTR 24

__global__ __launch_bounds__(256)
void glifr_gemm(const float* __restrict__ x, const float* __restrict__ fd,
                const float* __restrict__ wiv, const float* __restrict__ wlat,
                float* __restrict__ syn) {
  __shared__ bf16 Ahi[BM * ASTR], Alo[BM * ASTR];
  __shared__ bf16 Bhi[256 * BSTR], Blo[256 * BSTR];

  const int t    = threadIdx.x;
  const int lane = t & 31;
  const int half = lane >> 4;
  const int r    = lane & 15;
  const int w    = t >> 5, wm = w >> 2, wn = w & 3;
  const int m0   = blockIdx.y * BM;
  const int n0   = blockIdx.x * BN;

  v8f acc[4][2] = {};
  const int KT = 2 * (INDIM / BK);

  for (int kt = 0; kt < KT; ++kt) {
    const float* Ag = (kt < INDIM / BK) ? x   : fd;
    const float* Bg = (kt < INDIM / BK) ? wiv : wlat;
    const int kk = (kt & (INDIM / BK - 1)) * BK;

    f32x4 av[4], bv[4];
    int am[4], ak[4], bk[4], bn[4];
#pragma unroll
    for (int i = 0; i < 4; ++i) {
      int c = t + i * 256;
      am[i] = c >> 3;  ak[i] = (c & 7) * 4;
      av[i] = *(const f32x4*)(Ag + (size_t)(m0 + am[i]) * INDIM + kk + ak[i]);
      bk[i] = c >> 5;  bn[i] = (c & 31) * 4;
      bv[i] = *(const f32x4*)(Bg + (size_t)(kk + bk[i]) * HDIM + n0 + bn[i]);
    }
    __syncthreads();
#pragma unroll
    for (int i = 0; i < 4; ++i) {
      bf16x4 h, l;
#pragma unroll
      for (int e = 0; e < 4; ++e) {
        float a  = av[i][e];
        bf16  ah = (bf16)a;
        h[e] = ah;  l[e] = (bf16)(a - (float)ah);
      }
      *(bf16x4*)&Ahi[am[i] * ASTR + ak[i]] = h;
      *(bf16x4*)&Alo[am[i] * ASTR + ak[i]] = l;
#pragma unroll
      for (int e = 0; e < 4; ++e) {
        float bxv = bv[i][e];
        bf16  bh  = (bf16)bxv;
        int nn  = bn[i] + e;
        int row = (nn >> 4) * 32 + (bk[i] >> 4) * 16 + (nn & 15);
        int pos = row * BSTR + (bk[i] & 15);
        Bhi[pos] = bh;
        Blo[pos] = (bf16)(bxv - (float)bh);
      }
    }
    __syncthreads();

    Frag fah[4], fal[4], fbh[2], fbl[2];
#pragma unroll
    for (int mi = 0; mi < 4; ++mi) {
      const bf16* p = &Ahi[(wm * 64 + mi * 16 + r) * ASTR + half * 16];
      fah[mi].h[0] = *(const bf16x8*)p;
      fah[mi].h[1] = *(const bf16x8*)(p + 8);
      const bf16* q = &Alo[(wm * 64 + mi * 16 + r) * ASTR + half * 16];
      fal[mi].h[0] = *(const bf16x8*)q;
      fal[mi].h[1] = *(const bf16x8*)(q + 8);
    }
#pragma unroll
    for (int ni = 0; ni < 2; ++ni) {
      int nt = wn * 2 + ni;
      const bf16* p = &Bhi[(nt * 32 + half * 16 + r) * BSTR];
      fbh[ni].h[0] = *(const bf16x8*)p;
      fbh[ni].h[1] = *(const bf16x8*)(p + 8);
      const bf16* q = &Blo[(nt * 32 + half * 16 + r) * BSTR];
      fbl[ni].h[0] = *(const bf16x8*)q;
      fbl[ni].h[1] = *(const bf16x8*)(q + 8);
    }
#pragma unroll
    for (int mi = 0; mi < 4; ++mi)
#pragma unroll
      for (int ni = 0; ni < 2; ++ni) {
        acc[mi][ni] = __builtin_amdgcn_wmma_f32_16x16x32_bf16(
            false, fah[mi].v, false, fbh[ni].v, (short)0, acc[mi][ni], false, false);
        acc[mi][ni] = __builtin_amdgcn_wmma_f32_16x16x32_bf16(
            false, fal[mi].v, false, fbh[ni].v, (short)0, acc[mi][ni], false, false);
        acc[mi][ni] = __builtin_amdgcn_wmma_f32_16x16x32_bf16(
            false, fah[mi].v, false, fbl[ni].v, (short)0, acc[mi][ni], false, false);
      }
  }
#pragma unroll
  for (int mi = 0; mi < 4; ++mi)
#pragma unroll
    for (int ni = 0; ni < 2; ++ni) {
      int row = m0 + wm * 64 + mi * 16 + half * 8;
      int col = n0 + wn * 32 + ni * 16 + r;
#pragma unroll
      for (int v = 0; v < 8; ++v)
        syn[(size_t)(row + v) * HDIM + col] = acc[mi][ni][v];
    }
}

// ---------------------------------------------------------------------------
// Elementwise GLIF state update (float4 vectorized).
// ---------------------------------------------------------------------------
__global__ __launch_bounds__(256)
void glifr_pointwise(const float* __restrict__ firing, const float* __restrict__ voltage,
                     const float* __restrict__ asc,
                     const float* __restrict__ thresh, const float* __restrict__ km_t,
                     const float* __restrict__ kj_t, const float* __restrict__ rj_t,
                     const float* __restrict__ aj, float* __restrict__ out) {
  const int idx = (blockIdx.x * 256 + threadIdx.x) * 4;
  const int h   = idx & (HDIM - 1);
  const float* syn = out + (size_t)4 * BH;

  f32x4 sc  = *(const f32x4*)(syn + idx);
  f32x4 f   = *(const f32x4*)(firing + idx);
  f32x4 vo  = *(const f32x4*)(voltage + idx);
  f32x4 a0  = *(const f32x4*)(asc + idx);
  f32x4 a1  = *(const f32x4*)(asc + (size_t)BH + idx);
  f32x4 th  = *(const f32x4*)(thresh + h);
  f32x4 km  = *(const f32x4*)(km_t + h);
  f32x4 kj0 = *(const f32x4*)(kj_t + h);
  f32x4 kj1 = *(const f32x4*)(kj_t + HDIM + h);
  f32x4 rj0 = *(const f32x4*)(rj_t + h);
  f32x4 rj1 = *(const f32x4*)(rj_t + HDIM + h);
  f32x4 aj0 = *(const f32x4*)(aj + h);
  f32x4 aj1 = *(const f32x4*)(aj + HDIM + h);

  f32x4 fn, vn, a0n, a1n;
#pragma unroll
  for (int e = 0; e < 4; ++e) {
    float sr0 = 1.0f - 2.0f * sigf(rj0[e]);
    float sr1 = 1.0f - 2.0f * sigf(rj1[e]);
    float dk0 = sigf(kj0[e]);                 // DT * k_j[0]
    float dk1 = sigf(kj1[e]);                 // DT * k_j[1]
    float ft  = f[e] * (DT_C / TAU_C);
    a0n[e] = (a0[e] * sr0 + aj0[e]) * ft + (1.0f - dk0) * a0[e];
    a1n[e] = (a1[e] * sr1 + aj1[e]) * ft + (1.0f - dk1) * a1[e];
    float skm = sigf(km[e]);                  // DT * k_m
    vn[e] = sc[e] + skm * R_C * (a0n[e] + a1n[e] + I0_C)
          + (1.0f - skm) * vo[e]
          - ft * vo[e];                       // V_RESET = 0
    fn[e] = sigf(vn[e] - th[e]);              // SIGMA_V = 1
  }
  *(f32x4*)(out + idx)                  = fn;
  *(f32x4*)(out + (size_t)BH + idx)     = vn;
  *(f32x4*)(out + (size_t)2 * BH + idx) = a0n;
  *(f32x4*)(out + (size_t)3 * BH + idx) = a1n;
}

// ---------------------------------------------------------------------------
extern "C" void kernel_launch(void* const* d_in, const int* in_sizes, int n_in,
                              void* d_out, int out_size, void* d_ws, size_t ws_size,
                              hipStream_t stream) {
  (void)in_sizes; (void)n_in; (void)out_size;
  const float* x    = (const float*)d_in[0];
  const float* fir  = (const float*)d_in[1];
  const float* vol  = (const float*)d_in[2];
  const float* asc  = (const float*)d_in[3];
  const float* fd   = (const float*)d_in[5];
  const float* wiv  = (const float*)d_in[6];
  const float* wlat = (const float*)d_in[7];
  const float* th   = (const float*)d_in[8];
  const float* km   = (const float*)d_in[9];
  const float* kj   = (const float*)d_in[10];
  const float* rj   = (const float*)d_in[11];
  const float* aj   = (const float*)d_in[12];
  float* out = (float*)d_out;
  float* syn = out + (size_t)4 * BH;

  const bool use_pk = ws_size >= PKMAT * 8 * sizeof(bf16);   // 64 MB
  dim3 gg(HDIM / BN, BDIM / BM);

  if (use_pk) {
    bf16* wsp = (bf16*)d_ws;
    bf16* xhi    = wsp + 0 * PKMAT;  bf16* xlo    = wsp + 1 * PKMAT;
    bf16* fdhi   = wsp + 2 * PKMAT;  bf16* fdlo   = wsp + 3 * PKMAT;
    bf16* wivhi  = wsp + 4 * PKMAT;  bf16* wivlo  = wsp + 5 * PKMAT;
    bf16* wlathi = wsp + 6 * PKMAT;  bf16* wlatlo = wsp + 7 * PKMAT;

    pack_a<<<1024, 256, 0, stream>>>(x,    xhi,    xlo);
    pack_a<<<1024, 256, 0, stream>>>(fd,   fdhi,   fdlo);
    pack_b<<<1024, 256, 0, stream>>>(wiv,  wivhi,  wivlo);
    pack_b<<<1024, 256, 0, stream>>>(wlat, wlathi, wlatlo);
    glifr_gemm_pk<<<gg, 256, 0, stream>>>(xhi, xlo, fdhi, fdlo,
                                          wivhi, wivlo, wlathi, wlatlo, syn);
  } else {
    glifr_gemm<<<gg, 256, 0, stream>>>(x, fd, wiv, wlat, syn);
  }
  glifr_pointwise<<<BH / (256 * 4), 256, 0, stream>>>(fir, vol, asc, th, km, kj,
                                                      rj, aj, out);
}
